// DRGNNetwork_27212912788251
// MI455X (gfx1250) — compile-verified, matrix operand order checked by reading, not run on hardware
//
#include <hip/hip_runtime.h>

typedef __attribute__((ext_vector_type(16))) __bf16 v16bf;
typedef __attribute__((ext_vector_type(8)))  float  v8f;

#define N_ANT 64
#define HID   128
#define NHEAD 4
#define DH    32
#define NEGV  9000000000000000.0f

// element strides (multiples of 8 -> 16B-aligned ds_load_b128 fragments)
#define SA   136   // activation buffers [64][SA] bf16
#define SW   136   // weight buffer [128][SW] bf16
#define SVT  72    // per-head transposed V [32][SVT] bf16
#define SATT 72    // attention weights [64][SATT] bf16
#define SSC  68    // scores [64][SSC] f32

// dynamic-LDS layout (byte offsets, all 16B aligned).  157184 B total ->
// two workgroups fit in the 320KB WGP LDS pool.
#define OFF_WL   0                             // 128*136*2 = 34816
#define OFF_ACTA (OFF_WL   + 128*SW*2)         // 34816
#define OFF_ACTB (OFF_ACTA + 64*SA*2)          // 52224
#define OFF_QB   (OFF_ACTB + 64*SA*2)          // 69632
#define OFF_KB   (OFF_QB + 64*SA*2)            // 87040
#define OFF_VT   (OFF_KB + 64*SA*2)            // 104448 (32*SVT*2 = 4608)
#define OFF_SC   (OFF_VT + 32*SVT*2)           // 109056 (64*SSC*4 = 17408)
#define OFF_ATT  (OFF_SC + 64*SSC*4)           // 126464 (64*SATT*2 = 9216)
#define OFF_AO   (OFF_ATT + 64*SATT*2)         // 135680 (64*SA*2 = 17408)
#define OFF_MB   (OFF_AO + 64*SA*2)            // 153088 (64*64 = 4096)
#define SMEM_BYTES (OFF_MB + 64*64)            // 157184
// GRU-phase overlays (QB/KB/VT/SC unused then)
#define OFF_HB   OFF_QB
#define OFF_RB   (OFF_HB + 64*SA*2)
#define OFF_ZB   (OFF_RB + 64*HID*2)

#define SMEM extern __shared__ char smem[]

__device__ __forceinline__ void st_bf(int base, int idx, __bf16 v) {
  SMEM; ((__bf16*)(smem + base))[idx] = v;
}
__device__ __forceinline__ __bf16 ld_bf(int base, int idx) {
  SMEM; return ((const __bf16*)(smem + base))[idx];
}
__device__ __forceinline__ void st_u16(int base, int idx, unsigned short v) {
  SMEM; ((unsigned short*)(smem + base))[idx] = v;
}
__device__ __forceinline__ void st_u32(int base, int idx, unsigned v) {
  SMEM; ((unsigned*)(smem + base))[idx] = v;
}
__device__ __forceinline__ void st_u128(int base, int idx, uint4 v) {
  SMEM; ((uint4*)(smem + base))[idx] = v;
}
__device__ __forceinline__ void st_f32(int base, int idx, float v) {
  SMEM; ((float*)(smem + base))[idx] = v;
}
__device__ __forceinline__ float ld_f32(int base, int idx) {
  SMEM; return ((const float*)(smem + base))[idx];
}
__device__ __forceinline__ void st_u8(int base, int idx, unsigned char v) {
  SMEM; ((unsigned char*)(smem + base))[idx] = v;
}
__device__ __forceinline__ unsigned char ld_u8(int base, int idx) {
  SMEM; return ((const unsigned char*)(smem + base))[idx];
}

// two f32 -> packed bf16x2 dword (one v_cvt_pk_bf16_f32)
__device__ __forceinline__ unsigned pack_bf2(float lo, float hi) {
  union { __bf16 h[2]; unsigned u; } x;
  x.h[0] = (__bf16)lo; x.h[1] = (__bf16)hi;
  return x.u;
}
// store a packed pair in the same column, rows idx0 and idx0+strideE:
// 1 packed cvt + ds_store_b16 (lo half) + ds_store_b16_d16_hi (hi half)
__device__ __forceinline__ void st_bf2rows(int base, int idx0, int strideE,
                                           float a, float b) {
  unsigned pk = pack_bf2(a, b);
  st_u16(base, idx0,           (unsigned short)(pk & 0xffffu));
  st_u16(base, idx0 + strideE, (unsigned short)(pk >> 16));
}

// A/B fragment for v_wmma_f32_16x16x32_bf16 (two 16B ds loads per fragment).
__device__ __forceinline__ v16bf load_frag(int base, int stride, int row0, int k0,
                                           int lane) {
  SMEM;
  int m = lane & 15, kh = (lane >> 4) & 1;
  const __bf16* p =
      (const __bf16*)(smem + base) + (row0 + m) * stride + k0 + (kh << 3);
  union { v16bf v; uint4 u[2]; } f;
  f.u[0] = *(const uint4*)p;
  f.u[1] = *(const uint4*)(p + 16);
  return f.v;
}

__device__ __forceinline__ v8f wmma_bf16(v16bf a, v16bf b, v8f c) {
  return __builtin_amdgcn_wmma_f32_16x16x32_bf16(false, a, false, b, (short)0, c,
                                                 false, false);
}

// ---- weight staging into OFF_WL (stride SW) ----
template <int ROWS, int COLS>
__device__ __forceinline__ void load_w_f32(const float* W, int tid) {
  constexpr int NV = ROWS * COLS / 4;
#pragma unroll 2
  for (int i = tid; i < NV; i += 256) {
    float4 v = ((const float4*)W)[i];
    int e = i * 4;
    int u = ((e / COLS) * SW + (e % COLS)) >> 1;
    st_u32(OFF_WL, u,     pack_bf2(v.x, v.y));
    st_u32(OFF_WL, u + 1, pack_bf2(v.z, v.w));
  }
}
template <int ROWS, int COLS>
__device__ __forceinline__ void load_w_bf(const unsigned short* W, int tid) {
  constexpr int NV = ROWS * COLS / 8;
#pragma unroll 2
  for (int i = tid; i < NV; i += 256) {
    uint4 v = ((const uint4*)W)[i];
    int e = i * 8;
    st_u128(OFF_WL, ((e / COLS) * SW + (e % COLS)) >> 3, v);
  }
}
template <bool PRE, int ROWS, int COLS>
__device__ __forceinline__ void stage_w(const void* W, int tid) {
  if (PRE) load_w_bf<ROWS, COLS>((const unsigned short*)W, tid);
  else     load_w_f32<ROWS, COLS>((const float*)W, tid);
}

// Out(MxN) = act?( A(MxK) @ W^T + bias ).  A at aBase (stride SA), W at OFF_WL.
template <int M, int N, int K, bool RELU, bool TRANSP>
__device__ __forceinline__ void wave_gemm(int aBase, int outBase, int so,
                                          const float* bias, int wave, int lane) {
  constexpr int TN = N >> 4;
  constexpr int TOTAL = (M >> 4) * TN;
#pragma unroll
  for (int t0 = 0; t0 < TOTAL; t0 += 8) {
    int t = t0 + wave;
    int mt = t / TN, nt = t % TN;
    v8f c = {};
#pragma unroll
    for (int kk = 0; kk < K; kk += 32)
      c = wmma_bf16(load_frag(aBase, SA, mt * 16, kk, lane),
                    load_frag(OFF_WL, SW, nt * 16, kk, lane), c);
    int col   = nt * 16 + (lane & 15);
    int rbase = mt * 16 + ((lane >> 4) << 3);
    float bv = bias[col];
#pragma unroll
    for (int r = 0; r < 8; r += 2) {
      float v0 = c[r] + bv, v1 = c[r + 1] + bv;
      if (RELU) { v0 = fmaxf(v0, 0.0f); v1 = fmaxf(v1, 0.0f); }
      if (TRANSP)  // rows consecutive in memory -> single b32 store
        st_u32(outBase, (col * so + rbase + r) >> 1, pack_bf2(v0, v1));
      else
        st_bf2rows(outBase, (rbase + r) * so + col, so, v0, v1);
    }
  }
}

template <bool PRE>
__device__ __forceinline__ void mha(int inBase, int outBase,
                                    const void* wq, const float* bq,
                                    const void* wk, const float* bk,
                                    const void* wv, const float* bv,
                                    const void* wo, const float* bo, int tid) {
  int wave = tid >> 5, lane = tid & 31;

  stage_w<PRE, HID, HID>(wq, tid); __syncthreads();
  wave_gemm<N_ANT, HID, HID, true, false>(inBase, OFF_QB, SA, bq, wave, lane);
  __syncthreads();
  stage_w<PRE, HID, HID>(wk, tid); __syncthreads();
  wave_gemm<N_ANT, HID, HID, true, false>(inBase, OFF_KB, SA, bk, wave, lane);
  __syncthreads();
  stage_w<PRE, HID, HID>(wv, tid); __syncthreads();   // wv stays staged all heads

  for (int h = 0; h < NHEAD; ++h) {
    {  // V_h slice (64x32, K=128) -> transposed VTh[d_local][m]; 1 tile/wave
      int mt = wave >> 1, nt = wave & 1;
      v8f c = {};
#pragma unroll
      for (int kk = 0; kk < HID; kk += 32)
        c = wmma_bf16(load_frag(inBase, SA, mt * 16, kk, lane),
                      load_frag(OFF_WL, SW, h * DH + nt * 16, kk, lane), c);
      int col   = nt * 16 + (lane & 15);               // d_local
      int rbase = mt * 16 + ((lane >> 4) << 3);        // m
      float bvv = bv[h * DH + col];
#pragma unroll
      for (int r = 0; r < 8; r += 2) {
        float v0 = fmaxf(c[r] + bvv, 0.0f), v1 = fmaxf(c[r + 1] + bvv, 0.0f);
        st_u32(OFF_VT, (col * SVT + rbase + r) >> 1, pack_bf2(v0, v1));
      }
    }
    // scores = (Q_h K_h^T)/sqrt(dh), masked select.  16 tiles -> 2 per wave.
#pragma unroll
    for (int i = 0; i < 2; ++i) {
      int t = wave * 2 + i;
      int mt = t >> 2, nt = t & 3;
      v8f c = {};
      c = wmma_bf16(load_frag(OFF_QB, SA, mt * 16, h * DH, lane),
                    load_frag(OFF_KB, SA, nt * 16, h * DH, lane), c);
      int col   = nt * 16 + (lane & 15);
      int rbase = mt * 16 + ((lane >> 4) << 3);
#pragma unroll
      for (int r = 0; r < 8; ++r) {
        int row = rbase + r;
        float s = c[r] * 0.1767766952966369f;          // 1/sqrt(32)
        st_f32(OFF_SC, row * SSC + col,                // m ? s : -NEG
               ld_u8(OFF_MB, row * N_ANT + col) ? s : -NEGV);
      }
    }
    __syncthreads();
    if (tid < N_ANT) {                                 // softmax, row per thread
      float mx = -3.4e38f;
      for (int j = 0; j < N_ANT; ++j) mx = fmaxf(mx, ld_f32(OFF_SC, tid * SSC + j));
      float sum = 0.0f;
      for (int j = 0; j < N_ANT; ++j) {
        float e = __expf(ld_f32(OFF_SC, tid * SSC + j) - mx);
        st_f32(OFF_SC, tid * SSC + j, e);
        sum += e;
      }
      float inv = 1.0f / sum;
      for (int j = 0; j < N_ANT; j += 2)
        st_u32(OFF_ATT, (tid * SATT + j) >> 1,
               pack_bf2(ld_f32(OFF_SC, tid * SSC + j) * inv,
                        ld_f32(OFF_SC, tid * SSC + j + 1) * inv));
    }
    __syncthreads();
    {  // out_h = att @ V_h : 8 tiles -> 1 per wave, K=64
      int mt = wave >> 1, nt = wave & 1;
      v8f c = {};
#pragma unroll
      for (int kk = 0; kk < N_ANT; kk += 32)
        c = wmma_bf16(load_frag(OFF_ATT, SATT, mt * 16, kk, lane),
                      load_frag(OFF_VT, SVT, nt * 16, kk, lane), c);
      int col   = h * DH + nt * 16 + (lane & 15);
      int rbase = mt * 16 + ((lane >> 4) << 3);
#pragma unroll
      for (int r = 0; r < 8; r += 2)
        st_bf2rows(OFF_AO, (rbase + r) * SA + col, SA, c[r], c[r + 1]);
    }
    __syncthreads();
  }
  stage_w<PRE, HID, HID>(wo, tid); __syncthreads();
  wave_gemm<N_ANT, HID, HID, true, false>(OFF_AO, outBase, SA, bo, wave, lane);
  __syncthreads();
}

struct Params {
  const float* x; const int* mask; const float* hidden;
  const void* enc_w; const float* enc_b;
  const void* q1_w; const float* q1_b; const void* k1_w; const float* k1_b;
  const void* v1_w; const float* v1_b; const void* o1_w; const float* o1_b;
  const void* q2_w; const float* q2_b; const void* k2_w; const float* k2_b;
  const void* v2_w; const float* v2_b; const void* o2_w; const float* o2_b;
  const void* gru_w_ih; const void* gru_w_hh;
  const float* gru_b_ih; const float* gru_b_hh;
  const void* lin_w; const float* lin_b;
  float* qs; float* h3;
};

template <bool PRE>
__device__ __forceinline__ const void* woff(const void* w, int elems) {
  if (PRE) return (const void*)((const unsigned short*)w + elems);
  return (const void*)((const float*)w + elems);
}

template <bool PRE>
__global__ __launch_bounds__(256) void drgn_fused_kernel(Params p) {
  int b = blockIdx.x, tid = threadIdx.x;
  int wave = tid >> 5, lane = tid & 31;

  const float* x     = p.x      + (size_t)b * N_ANT * 64;
  const int*   maskp = p.mask   + (size_t)b * N_ANT * N_ANT;
  const float* hg    = p.hidden + (size_t)b * N_ANT * HID;

  // stage x (64x64 f32 -> bf16), mask (int -> byte)
#pragma unroll
  for (int i = tid; i < N_ANT * 64 / 4; i += 256) {
    float4 v = ((const float4*)x)[i];
    int e = i * 4;
    int u = ((e >> 6) * SA + (e & 63)) >> 1;
    st_u32(OFF_ACTA, u,     pack_bf2(v.x, v.y));
    st_u32(OFF_ACTA, u + 1, pack_bf2(v.z, v.w));
  }
#pragma unroll
  for (int i = tid; i < N_ANT * N_ANT; i += 256)
    st_u8(OFF_MB, i, (unsigned char)maskp[i]);
  stage_w<PRE, HID, 64>(p.enc_w, tid);
  __syncthreads();
  wave_gemm<N_ANT, HID, 64, true, false>(OFF_ACTA, OFF_ACTB, SA, p.enc_b, wave, lane);
  __syncthreads();

  mha<PRE>(OFF_ACTB, OFF_ACTA, p.q1_w, p.q1_b, p.k1_w, p.k1_b, p.v1_w, p.v1_b,
           p.o1_w, p.o1_b, tid);                        // h1 -> actA
  mha<PRE>(OFF_ACTA, OFF_ACTB, p.q2_w, p.q2_b, p.k2_w, p.k2_b, p.v2_w, p.v2_b,
           p.o2_w, p.o2_b, tid);                        // h2 -> actB

  // ---- GRU: stage hidden ----
#pragma unroll
  for (int i = tid; i < N_ANT * HID / 4; i += 256) {
    float4 v = ((const float4*)hg)[i];
    int e = i * 4;
    int u = ((e >> 7) * SA + (e & 127)) >> 1;
    st_u32(OFF_HB, u,     pack_bf2(v.x, v.y));
    st_u32(OFF_HB, u + 1, pack_bf2(v.z, v.w));
  }
  __syncthreads();

  v8f gi[4], gh[4];
  for (int c = 0; c < 3; ++c) {       // r, z, n gate chunks (each 128 wide)
    stage_w<PRE, HID, HID>(woff<PRE>(p.gru_w_ih, c * HID * HID), tid);
    __syncthreads();
#pragma unroll
    for (int i = 0; i < 4; ++i) {     // 32 tiles / 8 waves = 4 per wave, in regs
      int t = wave * 4 + i, mt = t >> 3, nt = t & 7;
      v8f acc = {};
#pragma unroll
      for (int kk = 0; kk < HID; kk += 32)
        acc = wmma_bf16(load_frag(OFF_ACTB, SA, mt * 16, kk, lane),
                        load_frag(OFF_WL, SW, nt * 16, kk, lane), acc);
      gi[i] = acc;
    }
    __syncthreads();
    stage_w<PRE, HID, HID>(woff<PRE>(p.gru_w_hh, c * HID * HID), tid);
    __syncthreads();
#pragma unroll
    for (int i = 0; i < 4; ++i) {
      int t = wave * 4 + i, mt = t >> 3, nt = t & 7;
      v8f acc = {};
#pragma unroll
      for (int kk = 0; kk < HID; kk += 32)
        acc = wmma_bf16(load_frag(OFF_HB, SA, mt * 16, kk, lane),
                        load_frag(OFF_WL, SW, nt * 16, kk, lane), acc);
      gh[i] = acc;
    }
#pragma unroll
    for (int i = 0; i < 4; ++i) {     // gate elementwise epilogue
      int t = wave * 4 + i, mt = t >> 3, nt = t & 7;
      int col   = nt * 16 + (lane & 15);
      int rbase = mt * 16 + ((lane >> 4) << 3);
      float bih = p.gru_b_ih[c * HID + col];
      float bhh = p.gru_b_hh[c * HID + col];
      if (c < 2) {
#pragma unroll
        for (int r = 0; r < 8; r += 2) {
          float g0 = 1.0f / (1.0f + __expf(-(gi[i][r] + bih + gh[i][r] + bhh)));
          float g1 = 1.0f / (1.0f + __expf(-(gi[i][r + 1] + bih + gh[i][r + 1] + bhh)));
          st_bf2rows(c == 0 ? OFF_RB : OFF_ZB, (rbase + 0 + r) * HID + col, HID, g0, g1);
        }
      } else {
#pragma unroll
        for (int r = 0; r < 8; ++r) {
          int row = rbase + r;
          float iv = gi[i][r] + bih;
          float hv = gh[i][r] + bhh;
          float rr = (float)ld_bf(OFF_RB, row * HID + col);
          float zz = (float)ld_bf(OFF_ZB, row * HID + col);
          float nn = tanhf(iv + rr * hv);
          float h3 = (1.0f - zz) * nn + zz * hg[row * HID + col];
          p.h3[(size_t)b * N_ANT * HID + row * HID + col] = h3;
          st_bf(OFF_ACTA, row * SA + col, (__bf16)h3);
        }
      }
    }
    __syncthreads();
  }

  // ---- head: qs = h3 @ lin_w^T + lin_b (N=20, padded to 32) ----
  stage_w<PRE, 20, HID>(p.lin_w, tid);
#pragma unroll
  for (int i = tid; i < 12 * HID / 2; i += 256) {        // zero-pad rows 20..31
    int e = i * 2;
    st_u32(OFF_WL, ((20 + e / HID) * SW + (e % HID)) >> 1, 0u);
  }
  __syncthreads();
  {
    int mt = wave >> 1, nt = wave & 1;                   // 8 tiles -> 1 per wave
    v8f acc = {};
#pragma unroll
    for (int kk = 0; kk < HID; kk += 32)
      acc = wmma_bf16(load_frag(OFF_ACTA, SA, mt * 16, kk, lane),
                      load_frag(OFF_WL, SW, nt * 16, kk, lane), acc);
    int col   = nt * 16 + (lane & 15);
    int rbase = mt * 16 + ((lane >> 4) << 3);
    if (col < 20) {
      float bv = p.lin_b[col];
#pragma unroll
      for (int r = 0; r < 8; ++r)
        p.qs[(size_t)b * N_ANT * 20 + (rbase + r) * 20 + col] = acc[r] + bv;
    }
  }
}

// one-shot f32 -> bf16 weight conversion into workspace
__global__ __launch_bounds__(256) void cvt_bf16_kernel(const float* __restrict__ s,
                                                       unsigned short* __restrict__ d,
                                                       int n4) {
  int i = blockIdx.x * 256 + threadIdx.x;
  if (i < n4) {
    float4 v = ((const float4*)s)[i];
    uint2 o; o.x = pack_bf2(v.x, v.y); o.y = pack_bf2(v.z, v.w);
    ((uint2*)d)[i] = o;
  }
}

extern "C" void kernel_launch(void* const* d_in, const int* in_sizes, int n_in,
                              void* d_out, int out_size, void* d_ws, size_t ws_size,
                              hipStream_t stream) {
  (void)n_in; (void)out_size;
  const float* wsrc[12] = {
    (const float*)d_in[3],
    (const float*)d_in[5],  (const float*)d_in[7],
    (const float*)d_in[9],  (const float*)d_in[11],
    (const float*)d_in[13], (const float*)d_in[15],
    (const float*)d_in[17], (const float*)d_in[19],
    (const float*)d_in[21], (const float*)d_in[22],
    (const float*)d_in[25]
  };
  const int wn[12] = {8192, 16384, 16384, 16384, 16384, 16384, 16384, 16384,
                      16384, 49152, 49152, 2560};
  size_t woffs[12]; size_t acc = 0;
  for (int i = 0; i < 12; ++i) { woffs[i] = acc; acc += wn[i]; }
  const size_t needWs = acc * 2;                             // 480256 bytes
  const bool pre = (d_ws != nullptr) && (ws_size >= needWs);

  unsigned short* wsb = (unsigned short*)d_ws;
  if (pre) {
    for (int i = 0; i < 12; ++i) {
      int n4 = wn[i] / 4;
      cvt_bf16_kernel<<<(n4 + 255) / 256, 256, 0, stream>>>(wsrc[i], wsb + woffs[i], n4);
    }
  }

  Params p;
  p.x      = (const float*)d_in[0];
  p.mask   = (const int*)  d_in[1];
  p.hidden = (const float*)d_in[2];
  auto W = [&](int slot) -> const void* {
    return pre ? (const void*)(wsb + woffs[slot]) : (const void*)wsrc[slot];
  };
  p.enc_w = W(0);  p.enc_b = (const float*)d_in[4];
  p.q1_w = W(1);  p.q1_b = (const float*)d_in[6];
  p.k1_w = W(2);  p.k1_b = (const float*)d_in[8];
  p.v1_w = W(3);  p.v1_b = (const float*)d_in[10];
  p.o1_w = W(4);  p.o1_b = (const float*)d_in[12];
  p.q2_w = W(5);  p.q2_b = (const float*)d_in[14];
  p.k2_w = W(6);  p.k2_b = (const float*)d_in[16];
  p.v2_w = W(7);  p.v2_b = (const float*)d_in[18];
  p.o2_w = W(8);  p.o2_b = (const float*)d_in[20];
  p.gru_w_ih = W(9);  p.gru_w_hh = W(10);
  p.gru_b_ih = (const float*)d_in[23]; p.gru_b_hh = (const float*)d_in[24];
  p.lin_w = W(11); p.lin_b = (const float*)d_in[26];

  int bs = in_sizes[0] / (N_ANT * 64);                       // 4096
  p.qs = (float*)d_out;
  p.h3 = (float*)d_out + (size_t)bs * N_ANT * 20;

  if (pre) {
    (void)hipFuncSetAttribute((const void*)drgn_fused_kernel<true>,
                              hipFuncAttributeMaxDynamicSharedMemorySize, SMEM_BYTES);
    drgn_fused_kernel<true><<<bs, 256, SMEM_BYTES, stream>>>(p);
  } else {
    (void)hipFuncSetAttribute((const void*)drgn_fused_kernel<false>,
                              hipFuncAttributeMaxDynamicSharedMemorySize, SMEM_BYTES);
    drgn_fused_kernel<false><<<bs, 256, SMEM_BYTES, stream>>>(p);
  }
}
